// MGDTModel_9698036155016
// MI455X (gfx1250) — matrix operand order repaired and checked by reading, not compiled, not run on hardware
//
#include <hip/hip_runtime.h>
#include <math.h>

// ---------------------------------------------------------------------------
// Types
// ---------------------------------------------------------------------------
typedef __bf16 bf16_t;
typedef __attribute__((ext_vector_type(16))) __bf16 v16bf;
typedef __attribute__((ext_vector_type(8)))  float  v8f;

__device__ __forceinline__ bf16_t f2bf(float f) {
  unsigned u = __builtin_bit_cast(unsigned, f);
  unsigned r = u + 0x7FFFu + ((u >> 16) & 1u);       // round-to-nearest-even
  unsigned short h = (unsigned short)(r >> 16);
  return __builtin_bit_cast(bf16_t, h);
}

__device__ __forceinline__ float gelu_exact(float x) {
  return 0.5f * x * (1.0f + erff(x * 0.70710678118654752f));
}

// CDNA5 async global->LDS copy (ASYNCcnt path, ISA 15.18.3 op 98).
// lds_off = addrspace(3) byte offset (low 32 bits of flat LDS pointer).
__device__ __forceinline__ void async_load_b128(unsigned lds_off, const void* gptr) {
  asm volatile("global_load_async_to_lds_b128 %0, %1, off"
               :: "v"(lds_off), "v"(gptr) : "memory");
}
__device__ __forceinline__ void wait_async0() {
  asm volatile("s_wait_asynccnt 0x0" ::: "memory");
}

// Model constants
#define MB   8
#define MT   16
#define MC   4
#define MH   84
#define MP   14
#define MNT  36
#define MTPS 39
#define ML   624          // T * TPS
#define MD   512
#define MNH  8
#define MDH  64
#define MNL  6
#define MDFF 2048
#define MNRET 120
#define MNACT 18
#define MNREW 3
#define ROWS (MB*ML)      // 4992
#define PK   800          // padded patch K (784 -> 800, multiple of 32)
#define PROWS (MB*MT*MNT) // 4608

// ---------------------------------------------------------------------------
// Generic tiled bf16 GEMM with WMMA:  C[M,N] = A[M,K] @ W[N,K]^T (+bias)(+res)
// flags: bit0 = accumulate into existing Cf (residual), bit1 = exact GELU
// ---------------------------------------------------------------------------
__global__ __launch_bounds__(128)
void gemm_bf16_wmma(const bf16_t* __restrict__ A, const bf16_t* __restrict__ W,
                    const float* __restrict__ bias,
                    float* __restrict__ Cf, bf16_t* __restrict__ Cb,
                    int M, int N, int K, int flags)
{
  __shared__ bf16_t As[64][40];
  __shared__ bf16_t Ws[64][40];

  const int tid  = threadIdx.x;
  const int lane = tid & 31;
  const int wid  = tid >> 5;
  const int hi   = lane >> 4;
  const int lm   = lane & 15;
  const int wm   = wid >> 1;     // wave row quadrant (0..1)
  const int wn   = wid & 1;      // wave col quadrant (0..1)
  const int m0   = blockIdx.y * 64;
  const int n0   = blockIdx.x * 64;
  const bool fullN = (n0 + 64 <= N);   // uniform: async fast path legal

  v8f acc[2][2] = {};

  const int lr = tid >> 1;            // staged row 0..63 (fallback path)
  const int lc = (tid & 1) * 16;      // staged col chunk (fallback path)

  for (int k0 = 0; k0 < K; k0 += 32) {
    if (fullN) {
      // ---- CDNA5 async global->LDS staging: 16B chunks, ASYNCcnt tracked ----
      // 64x32 bf16 tile = 256 chunks of 16B; 128 threads x 2 chunks each.
      if (k0 + 32 < K)
        __builtin_prefetch(A + (size_t)(m0 + lr) * K + k0 + 32, 0, 1);
      #pragma unroll
      for (int u = 0; u < 2; ++u) {
        const int cid = tid * 2 + u;       // 0..255
        const int r   = cid >> 2;          // row 0..63
        const int c8  = (cid & 3) * 8;     // bf16 col: 0,8,16,24
        async_load_b128((unsigned)(uintptr_t)&As[r][c8],
                        A + (size_t)(m0 + r) * K + k0 + c8);
        async_load_b128((unsigned)(uintptr_t)&Ws[r][c8],
                        W + (size_t)(n0 + r) * K + k0 + c8);
      }
      wait_async0();
    } else {
      // ---- guarded VGPR staging (N-edge tiles of the small head GEMMs) ----
      const bf16_t* srcA = A + (size_t)(m0 + lr) * K + k0 + lc;
      #pragma unroll
      for (int j = 0; j < 16; ++j) As[lr][lc + j] = srcA[j];

      const int nrow = n0 + lr;
      if (nrow < N) {
        const bf16_t* srcW = W + (size_t)nrow * K + k0 + lc;
        #pragma unroll
        for (int j = 0; j < 16; ++j) Ws[lr][lc + j] = srcW[j];
      } else {
        #pragma unroll
        for (int j = 0; j < 16; ++j) Ws[lr][lc + j] = f2bf(0.0f);
      }
    }
    __syncthreads();

    // ---- load fragments, k = 16*hi + j for both A and B ----
    v16bf af[2], wf[2];
    #pragma unroll
    for (int s = 0; s < 2; ++s) {
      const int mrow = wm * 32 + s * 16 + lm;      // A: lane holds row M
      #pragma unroll
      for (int j = 0; j < 16; ++j) {
        const int kk = hi * 16 + j;
        af[s][j] = As[mrow][kk];
      }
      const int ncol = wn * 32 + s * 16 + lm;      // B: lane holds col N
      #pragma unroll
      for (int j = 0; j < 16; ++j)
        wf[s][j] = Ws[ncol][hi * 16 + j];
    }

    #pragma unroll
    for (int i = 0; i < 2; ++i)
      #pragma unroll
      for (int j = 0; j < 2; ++j)
        acc[i][j] = __builtin_amdgcn_wmma_f32_16x16x32_bf16(
            false, af[i], false, wf[j], (short)0, acc[i][j], false, false);

    __syncthreads();
  }

  // ---- epilogue: bias / gelu / residual, f32 and/or bf16 stores ----
  #pragma unroll
  for (int i = 0; i < 2; ++i) {
    #pragma unroll
    for (int j = 0; j < 2; ++j) {
      #pragma unroll
      for (int r = 0; r < 8; ++r) {
        const int row = m0 + wm * 32 + i * 16 + hi * 8 + r;
        const int col = n0 + wn * 32 + j * 16 + lm;
        if (row < M && col < N) {
          float v = acc[i][j][r];
          if (bias)      v += bias[col];
          if (flags & 2) v  = gelu_exact(v);
          if (flags & 1) v += Cf[(size_t)row * N + col];
          if (Cf) Cf[(size_t)row * N + col] = v;
          if (Cb) Cb[(size_t)row * N + col] = f2bf(v);
        }
      }
    }
  }
}

// ---------------------------------------------------------------------------
// Fused flash-style attention (one wave per 16-query tile per (b,h))
// Q,K,V: bf16 [B,NH,L,DH]; O: bf16 [B,L,D]
// ---------------------------------------------------------------------------
__global__ __launch_bounds__(32)
void attn_wmma(const bf16_t* __restrict__ Q, const bf16_t* __restrict__ Kv,
               const bf16_t* __restrict__ Vv, bf16_t* __restrict__ O)
{
  const int qt   = blockIdx.x;        // 0..38
  const int bh   = blockIdx.y;        // 0..63
  const int b    = bh >> 3;
  const int h    = bh & 7;
  const int lane = threadIdx.x;
  const int hi   = lane >> 4;
  const int lm   = lane & 15;
  const size_t base = (size_t)bh * ML * MDH;

  __shared__ bf16_t pls[16][36];      // 16x32 P tile (+pad), C->A layout bounce

  // Q fragments: 16x64 as two 16x32 A-frags
  v16bf qa[2];
  #pragma unroll
  for (int c = 0; c < 2; ++c)
    #pragma unroll
    for (int j = 0; j < 16; ++j) {
      const int kk = c * 32 + hi * 16 + j;
      qa[c][j] = Q[base + (size_t)(qt * 16 + lm) * MDH + kk];
    }

  float mrun[8], lrun[8];
  v8f acc[4] = {};
  #pragma unroll
  for (int r = 0; r < 8; ++r) { mrun[r] = -1e30f; lrun[r] = 0.0f; }

  const float scale = 0.125f;         // 1/sqrt(64)

  for (int jt = 0; jt < 20; ++jt) {   // 20 key tiles of 32 (last partially masked)
    // ---- scores S = Q K^T for 16x32 keys ----
    v8f s[2] = {};
    #pragma unroll
    for (int sn = 0; sn < 2; ++sn) {
      const int kidx = jt * 32 + sn * 16 + lm;
      const bool kin = kidx < ML;
      #pragma unroll
      for (int c = 0; c < 2; ++c) {
        v16bf kb;
        #pragma unroll
        for (int j = 0; j < 16; ++j)
          kb[j] = kin ? Kv[base + (size_t)kidx * MDH + c * 32 + hi * 16 + j]
                      : f2bf(0.0f);
        s[sn] = __builtin_amdgcn_wmma_f32_16x16x32_bf16(
            false, qa[c], false, kb, (short)0, s[sn], false, false);
      }
    }

    // ---- mask + online softmax ----
    float p[2][8];
    #pragma unroll
    for (int r = 0; r < 8; ++r) {
      const int qrow = qt * 16 + hi * 8 + r;
      const bool qobs = (qrow % MTPS) < MNT;
      const int  qstep = qrow / MTPS;
      float best = -1e30f;
      #pragma unroll
      for (int sn = 0; sn < 2; ++sn) {
        const int key = jt * 32 + sn * 16 + lm;
        const bool kobs = (key % MTPS) < MNT;
        const bool ok = (key < ML) &&
                        ((key <= qrow) || (qobs && kobs && (key / MTPS) == qstep));
        const float v = ok ? s[sn][r] * scale : -1e30f;
        p[sn][r] = v;
        best = fmaxf(best, v);
      }
      #pragma unroll
      for (int mk = 1; mk < 16; mk <<= 1)
        best = fmaxf(best, __shfl_xor(best, mk, 32));

      const float mnew = fmaxf(mrun[r], best);
      const float corr = __expf(mrun[r] - mnew);
      mrun[r] = mnew;
      float rsum = 0.0f;
      #pragma unroll
      for (int sn = 0; sn < 2; ++sn) {
        const float e = __expf(p[sn][r] - mnew);
        p[sn][r] = e;
        rsum += e;
      }
      #pragma unroll
      for (int mk = 1; mk < 16; mk <<= 1)
        rsum += __shfl_xor(rsum, mk, 32);
      lrun[r] = lrun[r] * corr + rsum;
      #pragma unroll
      for (int nc = 0; nc < 4; ++nc) acc[nc][r] *= corr;
    }

    // ---- transpose P (C layout) -> A-frag layout through LDS ----
    __syncthreads();
    #pragma unroll
    for (int sn = 0; sn < 2; ++sn)
      #pragma unroll
      for (int r = 0; r < 8; ++r)
        pls[hi * 8 + r][sn * 16 + lm] = f2bf(p[sn][r]);
    __syncthreads();

    v16bf pa;
    #pragma unroll
    for (int j = 0; j < 16; ++j)
      pa[j] = pls[lm][hi * 16 + j];

    // ---- O += P @ V  (K=32 keys, N=64 dh as 4 chunks of 16) ----
    #pragma unroll
    for (int nc = 0; nc < 4; ++nc) {
      v16bf vb;
      #pragma unroll
      for (int j = 0; j < 16; ++j) {
        const int kidx = jt * 32 + hi * 16 + j;
        vb[j] = (kidx < ML) ? Vv[base + (size_t)kidx * MDH + nc * 16 + lm]
                            : f2bf(0.0f);
      }
      acc[nc] = __builtin_amdgcn_wmma_f32_16x16x32_bf16(
          false, pa, false, vb, (short)0, acc[nc], false, false);
    }
  }

  // ---- normalize + write [B,L,D] ----
  #pragma unroll
  for (int nc = 0; nc < 4; ++nc)
    #pragma unroll
    for (int r = 0; r < 8; ++r) {
      const int q = qt * 16 + hi * 8 + r;
      const float v = acc[nc][r] / lrun[r];
      O[((size_t)b * ML + q) * MD + h * MDH + nc * 16 + lm] = f2bf(v);
    }
}

// ---------------------------------------------------------------------------
// Elementwise / reduction helpers
// ---------------------------------------------------------------------------
__global__ __launch_bounds__(256)
void cvt_f32_bf16(const float* __restrict__ s, bf16_t* __restrict__ d, int n) {
  int i = blockIdx.x * 256 + threadIdx.x;
  if (i < n) d[i] = f2bf(s[i]);
}

__global__ __launch_bounds__(256)
void cvt_padK(const float* __restrict__ W, bf16_t* __restrict__ Wp,
              int N, int K, int Kp) {
  int i = blockIdx.x * 256 + threadIdx.x;
  if (i >= N * Kp) return;
  int col = i % Kp, row = i / Kp;
  Wp[i] = (col < K) ? f2bf(W[(size_t)row * K + col]) : f2bf(0.0f);
}

__global__ __launch_bounds__(256)
void im2col_patches(const float* __restrict__ frames, bf16_t* __restrict__ X) {
  int i = blockIdx.x * 256 + threadIdx.x;
  if (i >= PROWS * PK) return;
  int col = i % PK, row = i / PK;
  bf16_t v = f2bf(0.0f);
  if (col < MC * MP * MP) {
    int b = row / (MT * MNT), rem = row % (MT * MNT);
    int t = rem / MNT, p = rem % MNT;
    int gy = p / 6, gx = p % 6;
    int c = col / (MP * MP);
    int rr = (col % (MP * MP)) / MP, cc = col % MP;
    v = f2bf(frames[((((size_t)(b * MT + t) * MC) + c) * MH + gy * MP + rr) * MH
                    + gx * MP + cc]);
  }
  X[i] = v;
}

__global__ __launch_bounds__(256)
void assemble_seq(const float* __restrict__ obs, const int* __restrict__ rtg,
                  const int* __restrict__ act, const int* __restrict__ rew,
                  const int* __restrict__ gid,
                  const float* __restrict__ ret_e, const float* __restrict__ act_e,
                  const float* __restrict__ rew_e, const float* __restrict__ game_e,
                  const float* __restrict__ pos_e, const float* __restrict__ type_e,
                  float* __restrict__ seq) {
  int i = blockIdx.x * 256 + threadIdx.x;
  if (i >= ROWS * MD) return;
  int d = i & (MD - 1);
  int l = (i >> 9) % ML;
  int b = i / (ML * MD);
  int t = l / MTPS, o = l % MTPS;
  float v;
  if (o < MNT)       v = obs[(((size_t)(b * MT + t) * MNT) + o) * MD + d];
  else if (o == MNT) v = ret_e[(size_t)rtg[b * MT + t] * MD + d];
  else if (o == MNT + 1) v = act_e[(size_t)act[b * MT + t] * MD + d];
  else               v = rew_e[(size_t)rew[b * MT + t] * MD + d];
  const int type_id = (o < MNT) ? 0 : (o - (MNT - 1));
  v += pos_e[(size_t)l * MD + d] + type_e[(size_t)type_id * MD + d]
     + game_e[(size_t)gid[b] * MD + d];
  seq[i] = v;
}

__global__ __launch_bounds__(256)
void layernorm_bf16(const float* __restrict__ X, const float* __restrict__ s,
                    const float* __restrict__ bia, bf16_t* __restrict__ Y) {
  const int row = blockIdx.x;
  const float* x = X + (size_t)row * MD;
  __shared__ float red[256];
  const int tid = threadIdx.x;

  float ls = 0.0f;
  for (int i = tid; i < MD; i += 256) ls += x[i];
  red[tid] = ls; __syncthreads();
  for (int k = 128; k > 0; k >>= 1) { if (tid < k) red[tid] += red[tid + k]; __syncthreads(); }
  const float mu = red[0] * (1.0f / MD);
  __syncthreads();

  float lv = 0.0f;
  for (int i = tid; i < MD; i += 256) { float dd = x[i] - mu; lv += dd * dd; }
  red[tid] = lv; __syncthreads();
  for (int k = 128; k > 0; k >>= 1) { if (tid < k) red[tid] += red[tid + k]; __syncthreads(); }
  const float inv = rsqrtf(red[0] * (1.0f / MD) + 1e-5f);

  for (int i = tid; i < MD; i += 256)
    Y[(size_t)row * MD + i] = f2bf((x[i] - mu) * inv * s[i] + bia[i]);
}

__global__ __launch_bounds__(256)
void qkv_pack(const float* __restrict__ qkv, bf16_t* __restrict__ Q,
              bf16_t* __restrict__ K, bf16_t* __restrict__ V) {
  int i = blockIdx.x * 256 + threadIdx.x;
  if (i >= ROWS * MD) return;
  int d = i & (MD - 1);
  int l = (i >> 9) % ML;
  int b = i / (ML * MD);
  int h = d >> 6, dh = d & 63;
  size_t dst = (((size_t)(b * MNH + h) * ML) + l) * MDH + dh;
  size_t src = ((size_t)(b * ML) + l) * (3 * MD);
  Q[dst] = f2bf(qkv[src + d]);
  K[dst] = f2bf(qkv[src + MD + d]);
  V[dst] = f2bf(qkv[src + 2 * MD + d]);
}

__global__ __launch_bounds__(256)
void gather_heads(const float* __restrict__ seq, bf16_t* __restrict__ rh,
                  bf16_t* __restrict__ ah, bf16_t* __restrict__ wh) {
  int i = blockIdx.x * 256 + threadIdx.x;
  if (i >= MB * MT * MD) return;
  int d = i & (MD - 1);
  int row = i >> 9;
  int b = row >> 4, t = row & 15;
  size_t base = ((size_t)b * ML + t * MTPS) * MD + d;
  rh[i] = f2bf(seq[base + (size_t)(MNT - 1) * MD]);   // token 35
  ah[i] = f2bf(seq[base + (size_t)MNT * MD]);          // token 36
  wh[i] = f2bf(seq[base + (size_t)(MNT + 1) * MD]);    // token 37
}

// ---------------------------------------------------------------------------
// Host orchestration
// ---------------------------------------------------------------------------
static inline size_t aln(size_t x) { return (x + 255) & ~(size_t)255; }

extern "C" void kernel_launch(void* const* d_in, const int* in_sizes, int n_in,
                              void* d_out, int out_size, void* d_ws, size_t ws_size,
                              hipStream_t stream) {
  (void)in_sizes; (void)n_in; (void)out_size; (void)ws_size;

  const float* frames   = (const float*)d_in[0];
  const int*   rtg      = (const int*)d_in[1];
  const int*   actions  = (const int*)d_in[2];
  const int*   rewards  = (const int*)d_in[3];
  const int*   gameids  = (const int*)d_in[4];
  const float* patch_w  = (const float*)d_in[5];
  const float* patch_b  = (const float*)d_in[6];
  const float* ret_e    = (const float*)d_in[7];
  const float* act_e    = (const float*)d_in[8];
  const float* rew_e    = (const float*)d_in[9];
  const float* game_e   = (const float*)d_in[10];
  const float* pos_e    = (const float*)d_in[11];
  const float* type_e   = (const float*)d_in[12];
  const float* qkv_w    = (const float*)d_in[13];
  const float* qkv_b    = (const float*)d_in[14];
  const float* out_w    = (const float*)d_in[15];
  const float* out_b    = (const float*)d_in[16];
  const float* ln1_s    = (const float*)d_in[17];
  const float* ln1_b    = (const float*)d_in[18];
  const float* ln2_s    = (const float*)d_in[19];
  const float* ln2_b    = (const float*)d_in[20];
  const float* ffn_w1   = (const float*)d_in[21];
  const float* ffn_b1   = (const float*)d_in[22];
  const float* ffn_w2   = (const float*)d_in[23];
  const float* ffn_b2   = (const float*)d_in[24];
  const float* ret_hw   = (const float*)d_in[25];
  const float* ret_hb   = (const float*)d_in[26];
  const float* act_hw   = (const float*)d_in[27];
  const float* act_hb   = (const float*)d_in[28];
  const float* rew_hw   = (const float*)d_in[29];
  const float* rew_hb   = (const float*)d_in[30];

  char* ws = (char*)d_ws;
  size_t off = 0;
  auto alloc = [&](size_t bytes) -> char* { char* p = ws + off; off = aln(off + bytes); return p; };

  bf16_t* Xp    = (bf16_t*)alloc((size_t)PROWS * PK * 2);
  bf16_t* Wp    = (bf16_t*)alloc((size_t)MD * PK * 2);
  float*  obs   = (float*) alloc((size_t)PROWS * MD * 4);
  float*  seq   = (float*) alloc((size_t)ROWS * MD * 4);
  bf16_t* xn    = (bf16_t*)alloc((size_t)ROWS * MD * 2);
  float*  qkvb  = (float*) alloc((size_t)ROWS * 3 * MD * 4);
  bf16_t* wqkv  = (bf16_t*)alloc((size_t)MNL * 3 * MD * MD * 2);
  bf16_t* wout  = (bf16_t*)alloc((size_t)MNL * MD * MD * 2);
  bf16_t* wff1  = (bf16_t*)alloc((size_t)MNL * MDFF * MD * 2);
  bf16_t* wff2  = (bf16_t*)alloc((size_t)MNL * MD * MDFF * 2);
  bf16_t* Qb    = (bf16_t*)alloc((size_t)ROWS * MD * 2);
  bf16_t* Kb    = (bf16_t*)alloc((size_t)ROWS * MD * 2);
  bf16_t* Vb    = (bf16_t*)alloc((size_t)ROWS * MD * 2);
  bf16_t* attno = (bf16_t*)alloc((size_t)ROWS * MD * 2);
  bf16_t* hdn   = (bf16_t*)alloc((size_t)ROWS * MDFF * 2);
  bf16_t* rh    = (bf16_t*)alloc((size_t)MB * MT * MD * 2);
  bf16_t* ah    = (bf16_t*)alloc((size_t)MB * MT * MD * 2);
  bf16_t* wh    = (bf16_t*)alloc((size_t)MB * MT * MD * 2);
  bf16_t* wret  = (bf16_t*)alloc((size_t)MNRET * MD * 2);
  bf16_t* wact  = (bf16_t*)alloc((size_t)MNACT * MD * 2);
  bf16_t* wrew  = (bf16_t*)alloc((size_t)MNREW * MD * 2);

  auto grid1 = [](int n) { return dim3((unsigned)((n + 255) / 256)); };
  auto gemm = [&](const bf16_t* A, const bf16_t* W, const float* bias,
                  float* Cf, bf16_t* Cb, int M, int N, int K, int flags) {
    dim3 g((unsigned)((N + 63) / 64), (unsigned)((M + 63) / 64));
    gemm_bf16_wmma<<<g, dim3(128), 0, stream>>>(A, W, bias, Cf, Cb, M, N, K, flags);
  };

  // ---- weight conversions ----
  cvt_padK<<<grid1(MD * PK), 256, 0, stream>>>(patch_w, Wp, MD, MC * MP * MP, PK);
  cvt_f32_bf16<<<grid1(MNL * 3 * MD * MD), 256, 0, stream>>>(qkv_w,  wqkv, MNL * 3 * MD * MD);
  cvt_f32_bf16<<<grid1(MNL * MD * MD),     256, 0, stream>>>(out_w,  wout, MNL * MD * MD);
  cvt_f32_bf16<<<grid1(MNL * MDFF * MD),   256, 0, stream>>>(ffn_w1, wff1, MNL * MDFF * MD);
  cvt_f32_bf16<<<grid1(MNL * MD * MDFF),   256, 0, stream>>>(ffn_w2, wff2, MNL * MD * MDFF);
  cvt_f32_bf16<<<grid1(MNRET * MD), 256, 0, stream>>>(ret_hw, wret, MNRET * MD);
  cvt_f32_bf16<<<grid1(MNACT * MD), 256, 0, stream>>>(act_hw, wact, MNACT * MD);
  cvt_f32_bf16<<<grid1(MNREW * MD), 256, 0, stream>>>(rew_hw, wrew, MNREW * MD);

  // ---- patch embedding ----
  im2col_patches<<<grid1(PROWS * PK), 256, 0, stream>>>(frames, Xp);
  gemm(Xp, Wp, patch_b, obs, nullptr, PROWS, MD, PK, 0);

  // ---- token/positional assembly ----
  assemble_seq<<<grid1(ROWS * MD), 256, 0, stream>>>(
      obs, rtg, actions, rewards, gameids, ret_e, act_e, rew_e, game_e,
      pos_e, type_e, seq);

  // ---- transformer layers ----
  for (int i = 0; i < MNL; ++i) {
    layernorm_bf16<<<dim3(ROWS), 256, 0, stream>>>(seq, ln1_s + i * MD, ln1_b + i * MD, xn);
    gemm(xn, wqkv + (size_t)i * 3 * MD * MD, qkv_b + (size_t)i * 3 * MD,
         qkvb, nullptr, ROWS, 3 * MD, MD, 0);
    qkv_pack<<<grid1(ROWS * MD), 256, 0, stream>>>(qkvb, Qb, Kb, Vb);
    attn_wmma<<<dim3(ML / 16, MB * MNH), dim3(32), 0, stream>>>(Qb, Kb, Vb, attno);
    gemm(attno, wout + (size_t)i * MD * MD, out_b + (size_t)i * MD,
         seq, nullptr, ROWS, MD, MD, 1);                       // residual add
    layernorm_bf16<<<dim3(ROWS), 256, 0, stream>>>(seq, ln2_s + i * MD, ln2_b + i * MD, xn);
    gemm(xn, wff1 + (size_t)i * MDFF * MD, ffn_b1 + (size_t)i * MDFF,
         nullptr, hdn, ROWS, MDFF, MD, 2);                     // exact GELU -> bf16
    gemm(hdn, wff2 + (size_t)i * MD * MDFF, ffn_b2 + (size_t)i * MD,
         seq, nullptr, ROWS, MD, MDFF, 1);                     // residual add
  }

  // ---- heads ----
  gather_heads<<<grid1(MB * MT * MD), 256, 0, stream>>>(seq, rh, ah, wh);
  float* out = (float*)d_out;
  gemm(rh, wret, ret_hb, out,                        nullptr, MB * MT, MNRET, MD, 0);
  gemm(ah, wact, act_hb, out + MB * MT * MNRET,      nullptr, MB * MT, MNACT, MD, 0);
  gemm(wh, wrew, rew_hb, out + MB * MT * (MNRET + MNACT), nullptr, MB * MT, MNREW, MD, 0);
}
